// fused_classSR_3class_rcan_net_89404039233923
// MI455X (gfx1250) — matrix operand order repaired and checked by reading, not compiled, not run on hardware
//
// ClassSR-3class-RCAN for MI455X (gfx1250, wave32, WMMA).
// Every 3x3 conv = implicit GEMM on v_wmma_f32_16x16x32_f16 (fp16 A/B, fp32 acc).
// Per block (128 thr = 4 waves): 16 Cout x 128 pixel tile; each wave runs two
// 16x16 WMMA tiles sharing one A fragment. Per K-chunk(32):
//   - A tile (16x32 fp16 weights) arrives via TDM: tensor_load_to_lds with an
//     SGPR-built D# (TENSORcnt, s_wait_tensorcnt 0).
//   - bias tile arrives once per kernel via global_load_async_to_lds_b128
//     (ASYNCcnt), waited only before the epilogue.
//   - B tile is an im2col gather stored transposed lB[n][k] so each lane's
//     fragment is two contiguous ds_load_b128.
// Bias/ReLU/residual-add/PixelShuffle fused in the epilogue. Routing (argmax +
// stable compaction, cap=24) on-device; static shapes => graph-capture safe.
// Workspace requirement: ~88 MB (offsets below).

#include <hip/hip_runtime.h>
#include <cstdint>

typedef __attribute__((ext_vector_type(16))) _Float16 v16h;
typedef __attribute__((ext_vector_type(8)))  _Float16 v8h;
typedef __attribute__((ext_vector_type(8)))  float    v8f;
typedef __attribute__((ext_vector_type(4)))  unsigned u32x4;
typedef __attribute__((ext_vector_type(8)))  unsigned u32x8;

#define CFLAG_RELU 1
#define CFLAG_SHUF 2
#define CFLAG_RES  4

// ---------------------------------------------------------------- classifier
__global__ __launch_bounds__(256)
void k_classify(const float* __restrict__ x, const float* __restrict__ w1,
                const float* __restrict__ b1, const float* __restrict__ fw,
                const float* __restrict__ fb, int* __restrict__ route) {
  __shared__ float pooled[32];
  const int tid = threadIdx.x, img = blockIdx.x;
  if (tid < 32) pooled[tid] = 0.0f;
  __syncthreads();
  const float* xi = x + (size_t)img * 3072;
  for (int i = tid; i < 32 * 1024; i += 256) {
    int co = i >> 10, p = i & 1023;
    int y = p >> 5, xx = p & 31;
    float s = b1[co];
    for (int ci = 0; ci < 3; ++ci)
      for (int kh = 0; kh < 3; ++kh) {
        int iy = y + kh - 1;
        if (iy < 0 || iy > 31) continue;
        for (int kw = 0; kw < 3; ++kw) {
          int ix = xx + kw - 1;
          if (ix < 0 || ix > 31) continue;
          s += w1[((co * 3 + ci) * 3 + kh) * 3 + kw] *
               xi[(ci << 10) + (iy << 5) + ix] * (1.0f / 255.0f);
        }
      }
    atomicAdd(&pooled[co], fmaxf(s, 0.0f) * (1.0f / 1024.0f));  // ds_add_f32
  }
  __syncthreads();
  if (tid == 0) {
    float best = -1e30f; int bi = 0;
    for (int j = 0; j < 3; ++j) {
      float l = fb[j];
      for (int c = 0; c < 32; ++c) l += pooled[c] * fw[c * 3 + j];
      if (l > best) { best = l; bi = j; }
    }
    route[img] = bi;
  }
}

// stable compaction into per-expert index lists (argsort-equivalent order)
__global__ void k_compact(const int* __restrict__ route, int* __restrict__ cnt,
                          int* __restrict__ idxl) {
  if (threadIdx.x == 0 && blockIdx.x == 0) {
    int c[3] = {0, 0, 0};
    for (int i = 0; i < 48; ++i) {
      int r = route[i];
      if (c[r] < 24) idxl[r * 24 + c[r]] = i;
      c[r]++;
    }
    cnt[0] = c[0]; cnt[1] = c[1]; cnt[2] = c[2];
  }
}

__global__ void k_zero(float* __restrict__ p, int n) {
  for (int i = blockIdx.x * blockDim.x + threadIdx.x; i < n;
       i += gridDim.x * blockDim.x)
    p[i] = 0.0f;
}

// gather routed images + fused sub_mean (1x1 conv 3->3), fp32 -> fp16
__global__ void k_gather(const float* __restrict__ x, const int* __restrict__ idxl,
                         const int* __restrict__ cnt, const float* __restrict__ sw,
                         const float* __restrict__ sb, _Float16* __restrict__ dst) {
  const int n = cnt[0];
  for (int i = blockIdx.x * blockDim.x + threadIdx.x; i < 24 * 3072;
       i += gridDim.x * blockDim.x) {
    int slot = i / 3072;
    int rem = i - slot * 3072;
    int co = rem >> 10, p = rem & 1023;
    float v = 0.0f;
    if (slot < n) {
      int img = idxl[slot];
      const float* xp = x + (size_t)img * 3072 + p;
      v = sb[co];
      for (int ci = 0; ci < 3; ++ci) v += sw[co * 3 + ci] * xp[ci << 10];
    }
    dst[i] = (_Float16)v;
  }
}

// weight pack: OIHW fp32 -> fp16 [coPad][Kpad] (k = ci*9+kh*3+kw, contiguous per
// co row, zero padded => TDM tile reads always valid). Bias is zero-padded to
// coPad floats right after the weights so the async bias DMA never reads OOB.
__global__ void k_pack(const float* __restrict__ w, const float* __restrict__ b,
                       _Float16* __restrict__ dst, int Cout, int Ktot, int Kpad,
                       int coPad, int total) {
  float* biasPad = (float*)(dst + (size_t)coPad * Kpad);
  for (int i = blockIdx.x * blockDim.x + threadIdx.x; i < total;
       i += gridDim.x * blockDim.x) {
    int co = i / Kpad, k = i - co * Kpad;
    float v = (co < Cout && k < Ktot) ? w[(size_t)co * Ktot + k] : 0.0f;
    dst[i] = (_Float16)v;
    if (i < coPad) biasPad[i] = (i < Cout) ? b[i] : 0.0f;
  }
}

// -------------------------------------------- WMMA implicit-GEMM 3x3 conv
__global__ __launch_bounds__(128)
void k_conv_wmma(const _Float16* __restrict__ in, _Float16* __restrict__ out,
                 const _Float16* __restrict__ wpack, const float* __restrict__ biasPad,
                 const _Float16* __restrict__ resid,
                 int Cin, int Cout, int coPad, int wShift, int hw, int Kpad,
                 int flags) {
  __shared__ __align__(16) _Float16 lA[16 * 32];     // weight tile (TDM dest)
  __shared__ __align__(16) _Float16 lB[128 * 32];    // im2col tile, [n][k]
  __shared__ __align__(16) float    lBias[16];       // async-DMA'd bias tile

  const int tid = threadIdx.x;
  const int wave = tid >> 5;
  const int lane = tid & 31;
  const int img = blockIdx.z;
  const int coBase = blockIdx.y << 4;
  const int pixBase = blockIdx.x << 7;

  const int W = 1 << wShift;
  const int H = hw >> wShift;
  const int Ktot = Cin * 9;
  const _Float16* inImg = in + (size_t)img * Cin * hw;
  __builtin_prefetch(inImg, 0, 0);  // global_prefetch_b8

  const unsigned aBase = (unsigned)(uintptr_t)(&lA[0]);
  const unsigned biasBase = (unsigned)(uintptr_t)(&lBias[0]);

  // Bias tile: 64B via per-lane async DMA; waited only before the epilogue.
  if (wave == 0 && lane < 4) {
    unsigned long long src =
        (unsigned long long)(uintptr_t)(biasPad + coBase + (lane << 2));
    unsigned dst = biasBase + (lane << 4);
    asm volatile("global_load_async_to_lds_b128 %0, %1, off"
                 :: "v"(dst), "v"(src) : "memory");
  }

  v8f acc0 = {};
  v8f acc1 = {};

  // per-thread invariant im2col decomposition (stride-128 staging => kk fixed)
  const int kkFix = tid & 31;
  const int nStart = tid >> 5;

  for (int k0 = 0; k0 < Kpad; k0 += 32) {
    // ---- A tile via Tensor Data Mover: 2D tile 32 elem x 16 rows of fp16 ----
    if (wave == 0) {
      unsigned long long ga = (unsigned long long)(uintptr_t)(
          wpack + (size_t)coBase * Kpad + k0);
      u32x4 g0;
      g0[0] = 1u;                                   // count=1 (user mode)
      g0[1] = aBase;                                // lds_addr
      g0[2] = (unsigned)ga;                         // global_addr[31:0]
      g0[3] = (unsigned)(ga >> 32) | (2u << 30);    // addr[56:32] | type=2
      u32x8 g1;
      g1[0] = 1u << 16;                             // data_size=2B
      g1[1] = ((unsigned)Kpad & 0xFFFFu) << 16;     // tensor_dim0 lo
      g1[2] = ((unsigned)Kpad >> 16) |
              (((unsigned)coPad & 0xFFFFu) << 16);  // dim0 hi | tensor_dim1 lo
      g1[3] = ((unsigned)coPad >> 16) | (32u << 16);// dim1 hi | tile_dim0=32
      g1[4] = 16u;                                  // tile_dim1=16, tile_dim2=0
      g1[5] = (unsigned)Kpad;                       // tensor_dim0_stride lo32
      g1[6] = 0u;
      g1[7] = 0u;
      asm volatile("tensor_load_to_lds %0, %1" :: "s"(g0), "s"(g1) : "memory");
    }

    // ---- B tile: im2col gather (zero padding), transposed store lB[n][k] ----
    {
      const int k = k0 + kkFix;
      const bool kvalid = k < Ktot;
      const int ci = k / 9;
      const int t = k - ci * 9;
      const int t3 = t / 3;
      const int dy = t3 - 1, dx = t - t3 * 3 - 1;
      const _Float16* base = inImg + (size_t)ci * hw;
      for (int n = nStart; n < 128; n += 4) {
        int p = pixBase + n;
        int y = p >> wShift, x = p & (W - 1);
        int iy = y + dy, ix = x + dx;
        _Float16 v = (_Float16)0.0f;
        if (kvalid && iy >= 0 && iy < H && ix >= 0 && ix < W)
          v = base[(iy << wShift) + ix];
        lB[n * 32 + kkFix] = v;
      }
    }
    __builtin_amdgcn_s_wait_tensorcnt(0);  // TDM landed (no-op for waves 1-3)
    __syncthreads();

    // ---- fragments (gfx1250 f16 WMMA layouts, wave32) + 2 WMMAs/iter ----
    {
      const int m = lane & 15;
      const int hi = lane >> 4;
      v8h a0 = *(const v8h*)&lA[m * 32 + (hi << 4)];
      v8h a1 = *(const v8h*)&lA[m * 32 + (hi << 4) + 8];
      v16h afrag = __builtin_shufflevector(a0, a1, 0, 1, 2, 3, 4, 5, 6, 7,
                                           8, 9, 10, 11, 12, 13, 14, 15);
      const _Float16* bp0 = &lB[((wave << 5) + m) * 32 + (hi << 4)];
      v8h b00 = *(const v8h*)bp0;
      v8h b01 = *(const v8h*)(bp0 + 8);
      v16h bfrag0 = __builtin_shufflevector(b00, b01, 0, 1, 2, 3, 4, 5, 6, 7,
                                            8, 9, 10, 11, 12, 13, 14, 15);
      const _Float16* bp1 = bp0 + 16 * 32;
      v8h b10 = *(const v8h*)bp1;
      v8h b11 = *(const v8h*)(bp1 + 8);
      v16h bfrag1 = __builtin_shufflevector(b10, b11, 0, 1, 2, 3, 4, 5, 6, 7,
                                            8, 9, 10, 11, 12, 13, 14, 15);
      acc0 = __builtin_amdgcn_wmma_f32_16x16x32_f16(false, afrag, false, bfrag0,
                                                    (short)0, acc0, false, false);
      acc1 = __builtin_amdgcn_wmma_f32_16x16x32_f16(false, afrag, false, bfrag1,
                                                    (short)0, acc1, false, false);
    }
    __syncthreads();
  }

  // bias DMA completion (overlapped with the whole K-loop)
  asm volatile("s_wait_asynccnt 0" ::: "memory");
  __syncthreads();

  // epilogue: bias + ReLU + residual / fused PixelShuffle(2) store
  const int hi = lane >> 4;
  const int n0 = (wave << 5) + (lane & 15);
#pragma unroll
  for (int s = 0; s < 2; ++s) {
    const int p = pixBase + n0 + (s << 4);
    const int y = p >> wShift, x = p & (W - 1);
#pragma unroll
    for (int i = 0; i < 8; ++i) {
      int co = coBase + i + (hi << 3);
      if (co >= Cout) continue;
      float v = (s ? acc1[i] : acc0[i]) + lBias[i + (hi << 3)];
      if (flags & CFLAG_RELU) v = fmaxf(v, 0.0f);
      if (flags & CFLAG_SHUF) {
        int oc = co >> 2, c4 = co & 3;
        int oy = (y << 1) + (c4 >> 1);
        int ox = (x << 1) + (c4 & 1);
        size_t o = (size_t)img * (Cout >> 2) * (hw << 2) +
                   (size_t)oc * (hw << 2) + ((size_t)oy << (wShift + 1)) + ox;
        out[o] = (_Float16)v;
      } else {
        size_t o = (size_t)img * Cout * hw + (size_t)co * hw + p;
        if (flags & CFLAG_RES) v += (float)resid[o];
        out[o] = (_Float16)v;
      }
    }
  }
}

// ------------------------------------------- channel attention (CALayer)
__global__ __launch_bounds__(256)
void k_ca(const _Float16* __restrict__ b2, const float* __restrict__ cdw,
          const float* __restrict__ cdb, const float* __restrict__ cuw,
          const float* __restrict__ cub, float* __restrict__ scale,
          int f, int fr, int hw) {
  __shared__ float pooled[64];
  __shared__ float sq[8];
  const int img = blockIdx.x;
  const int wave = threadIdx.x >> 5, lane = threadIdx.x & 31;
  const _Float16* src = b2 + (size_t)img * f * hw;
  for (int c = wave; c < f; c += 8) {
    float s = 0.0f;
    for (int pp = lane; pp < hw; pp += 32) s += (float)src[(size_t)c * hw + pp];
    for (int off = 16; off; off >>= 1) s += __shfl_xor(s, off, 32);
    if (lane == 0) pooled[c] = s / (float)hw;
  }
  __syncthreads();
  if (threadIdx.x < fr) {
    float s = cdb[threadIdx.x];
    for (int c = 0; c < f; ++c) s += cdw[threadIdx.x * f + c] * pooled[c];
    sq[threadIdx.x] = fmaxf(s, 0.0f);
  }
  __syncthreads();
  for (int c = threadIdx.x; c < f; c += 256) {
    float s = cub[c];
    for (int j = 0; j < fr; ++j) s += cuw[c * fr + j] * sq[j];
    scale[(img << 6) + c] = 1.0f / (1.0f + expf(-s));
  }
}

// h_new = h + b2 * sigmoid-scale (RCAB tail)
__global__ void k_apply(const _Float16* __restrict__ h, const _Float16* __restrict__ b2,
                        const float* __restrict__ scale, _Float16* __restrict__ out,
                        int f, int total) {
  for (int i = blockIdx.x * blockDim.x + threadIdx.x; i < total;
       i += gridDim.x * blockDim.x) {
    int c = (i >> 10) % f;
    int img = i / (f << 10);
    out[i] = (_Float16)((float)h[i] + (float)b2[i] * scale[(img << 6) + c]);
  }
}

// add_mean (1x1 3->3) + masked scatter to routed output positions
__global__ void k_final(const _Float16* __restrict__ T, const int* __restrict__ idxl,
                        const int* __restrict__ cnt, const float* __restrict__ aw,
                        const float* __restrict__ ab, float* __restrict__ outp) {
  const int n = cnt[0];
  const int HW = 128 * 128;
  for (int i = blockIdx.x * blockDim.x + threadIdx.x; i < 24 * 3 * HW;
       i += gridDim.x * blockDim.x) {
    int slot = i / (3 * HW);
    if (slot >= n) continue;
    int rem = i - slot * 3 * HW;
    int co = rem / HW;
    int p = rem - co * HW;
    int img = idxl[slot];
    const _Float16* tp = T + (size_t)slot * 3 * HW + p;
    float v = ab[co];
    for (int ci = 0; ci < 3; ++ci)
      v += aw[co * 3 + ci] * (float)tp[(size_t)ci * HW];
    outp[(size_t)img * 3 * HW + (size_t)co * HW + p] = v;
  }
}

// ------------------------------------------------------------- host side
// d_in layout (setup_inputs dict insertion order):
//   0:x  1:clf.w1 2:clf.b1 3:clf.fw 4:clf.fb
//   expert e base EB=5+e*66: +0 sub_w +1 sub_b +2 head_w +3 head_b
//     group g base 4+g*26: block b base +b*8: w1,b1,w2,b2,cdw,cdb,cuw,cub ;
//       cw=+24 cb=+25
//     +56 btw +57 btb +58 u1w +59 u1b +60 u2w +61 u2b +62 tw +63 tb
//     +64 addw +65 addb
static const size_t OFF_ROUTE = 0;
static const size_t OFF_CNT   = 1024;
static const size_t OFF_IDX   = 2048;
static const size_t OFF_SCALE = 4096;           // 24*64 floats
static const size_t OFF_PACK  = 16384;          // 256*576 halves + 256 floats
static const size_t OFF_BUFX  = 0x60000;        // 24*3*1024 f16
static const size_t OFF_F0    = 0x100000;       // 6 feature buffers
static const size_t FEAT_STRIDE = 0x340000;
static const size_t OFF_P1    = 0x1500000;      // 24*64*4096 f16
static const size_t OFF_P2    = 0x2300000;      // 24*64*16384 f16
static const size_t OFF_T     = 0x5400000;      // 24*3*16384 f16

static void launch_conv(hipStream_t s, const _Float16* in, _Float16* out,
                        const float* w, const float* b, const _Float16* resid,
                        int Cin, int Cout, int wShift, int hw, int flags,
                        _Float16* packBuf) {
  int Ktot = Cin * 9;
  int Kpad = (Ktot + 31) & ~31;
  int coPad = (Cout + 15) & ~15;
  int total = coPad * Kpad;
  k_pack<<<(total + 255) / 256, 256, 0, s>>>(w, b, packBuf, Cout, Ktot, Kpad,
                                             coPad, total);
  const float* biasPad = (const float*)(packBuf + (size_t)coPad * Kpad);
  dim3 grid(hw >> 7, coPad >> 4, 24);
  k_conv_wmma<<<grid, 128, 0, s>>>(in, out, packBuf, biasPad, resid, Cin, Cout,
                                   coPad, wShift, hw, Kpad, flags);
}

extern "C" void kernel_launch(void* const* d_in, const int* in_sizes, int n_in,
                              void* d_out, int out_size, void* d_ws, size_t ws_size,
                              hipStream_t stream) {
  char* ws = (char*)d_ws;
  const float* x = (const float*)d_in[0];
  int* route = (int*)(ws + OFF_ROUTE);
  int* cnt   = (int*)(ws + OFF_CNT);
  int* idxl  = (int*)(ws + OFF_IDX);
  float* scale = (float*)(ws + OFF_SCALE);
  _Float16* packBuf = (_Float16*)(ws + OFF_PACK);
  _Float16* bufX = (_Float16*)(ws + OFF_BUFX);
  _Float16* FB[6];
  for (int i = 0; i < 6; ++i) FB[i] = (_Float16*)(ws + OFF_F0 + i * FEAT_STRIDE);
  _Float16* P1 = (_Float16*)(ws + OFF_P1);
  _Float16* P2 = (_Float16*)(ws + OFF_P2);
  _Float16* Tb = (_Float16*)(ws + OFF_T);

  k_classify<<<48, 256, 0, stream>>>(x, (const float*)d_in[1], (const float*)d_in[2],
                                     (const float*)d_in[3], (const float*)d_in[4], route);
  k_compact<<<1, 32, 0, stream>>>(route, cnt, idxl);
  k_zero<<<2048, 256, 0, stream>>>((float*)d_out, out_size);

  static const int feats[3] = {36, 50, 64};
  for (int e = 0; e < 3; ++e) {
    const int EB = 5 + e * 66;
    const int f = feats[e];
    const int fr = f / 16;
    const int* idxE = idxl + e * 24;
    const int* cntE = cnt + e;
    const size_t featBytes = (size_t)24 * f * 1024 * sizeof(_Float16);
    const int featElems = 24 * f * 1024;
#define EP(i) ((const float*)d_in[EB + (i)])

    _Float16* F0 = FB[0];  // head output (long skip, preserved)
    _Float16* H0 = FB[1];
    _Float16* H1 = FB[2];
    _Float16* Ab = FB[3];
    _Float16* Bb = FB[4];
    _Float16* GIN = FB[5];

    k_gather<<<(24 * 3072 + 255) / 256, 256, 0, stream>>>(x, idxE, cntE, EP(0), EP(1), bufX);
    launch_conv(stream, bufX, F0, EP(2), EP(3), nullptr, 3, f, 5, 1024, 0, packBuf);
    hipMemcpyAsync(H0, F0, featBytes, hipMemcpyDeviceToDevice, stream);

    _Float16* cur = H0;
    _Float16* alt = H1;
    for (int g = 0; g < 2; ++g) {
      const int GB = 4 + g * 26;
      hipMemcpyAsync(GIN, cur, featBytes, hipMemcpyDeviceToDevice, stream);
      for (int b = 0; b < 3; ++b) {
        const int BB = GB + b * 8;
        launch_conv(stream, cur, Ab, EP(BB + 0), EP(BB + 1), nullptr, f, f, 5, 1024,
                    CFLAG_RELU, packBuf);
        launch_conv(stream, Ab, Bb, EP(BB + 2), EP(BB + 3), nullptr, f, f, 5, 1024,
                    0, packBuf);
        k_ca<<<24, 256, 0, stream>>>(Bb, EP(BB + 4), EP(BB + 5), EP(BB + 6), EP(BB + 7),
                                     scale, f, fr, 1024);
        k_apply<<<(featElems + 255) / 256, 256, 0, stream>>>(cur, Bb, scale, alt, f, featElems);
        _Float16* t = cur; cur = alt; alt = t;
      }
      launch_conv(stream, cur, alt, EP(GB + 24), EP(GB + 25), GIN, f, f, 5, 1024,
                  CFLAG_RES, packBuf);
      _Float16* t = cur; cur = alt; alt = t;
    }
    launch_conv(stream, cur, alt, EP(56), EP(57), F0, f, f, 5, 1024, CFLAG_RES, packBuf);
    { _Float16* t = cur; cur = alt; alt = t; }
    launch_conv(stream, cur, P1, EP(58), EP(59), nullptr, f, 4 * f, 5, 1024,
                CFLAG_SHUF, packBuf);
    launch_conv(stream, P1, P2, EP(60), EP(61), nullptr, f, 4 * f, 6, 4096,
                CFLAG_SHUF, packBuf);
    launch_conv(stream, P2, Tb, EP(62), EP(63), nullptr, f, 3, 7, 16384, 0, packBuf);
    k_final<<<4608, 256, 0, stream>>>(Tb, idxE, cntE, EP(64), EP(65), (float*)d_out);
#undef EP
  }
  (void)in_sizes; (void)n_in; (void)ws_size;
}
